// RNN_custom_68693706932657
// MI455X (gfx1250) — matrix operand order repaired and checked by reading, not compiled, not run on hardware
//
#include <hip/hip_runtime.h>

// ---------------------------------------------------------------------------
// Spiking RNN forward (B=128, T=256, I=512, H=1024, D=512) for gfx1250.
// bf16 WMMA (v_wmma_f32_16x16x32_bf16) GEMMs, f32 accumulate, f32 LIF update.
// Async global->LDS staging for spike panels where the toolchain supports it.
// ---------------------------------------------------------------------------

typedef __attribute__((ext_vector_type(16))) __bf16 v16bf;
typedef __attribute__((ext_vector_type(8)))  __bf16 v8bf;
typedef __attribute__((ext_vector_type(8)))  float  v8f;

#define B_ 128
#define T_ 256
#define I_ 512
#define H_ 1024
#define D_ 512
#define BJ0 0.1f

#if __has_builtin(__builtin_amdgcn_global_load_async_to_lds_b128) && \
    __has_builtin(__builtin_amdgcn_s_wait_asynccnt)
#define ASYNC_LDS 1
// Builtin signature (recovered from hipcc diagnostics):
//   (v4i addrspace(1)* gsrc, v4i addrspace(3)* lds, int offset, int cpol)
typedef int v4i_ __attribute__((__vector_size__(16)));
typedef __attribute__((address_space(1))) v4i_* gv4p;
typedef __attribute__((address_space(3))) v4i_* sv4p;
#else
#define ASYNC_LDS 0
#endif

__device__ __forceinline__ unsigned short f2bf_u16(float f) {
  unsigned u = __builtin_bit_cast(unsigned, f);
  return (unsigned short)((u + 0x7FFFu + ((u >> 16) & 1u)) >> 16);  // RNE
}
__device__ __forceinline__ float bf2f(unsigned short h) {
  return __builtin_bit_cast(float, (unsigned)h << 16);
}

// A fragment (16x32 bf16): lane L holds row M=L&15, two 8-elem chunks
// (K = 16*(L>>4) and K = 16*(L>>4) + 8), matching the B fragment.
__device__ __forceinline__ v16bf load_a_frag(const unsigned short* arow, int k, int hi) {
  union { v16bf v; v8bf h[2]; } av;
  av.h[0] = *(const v8bf*)(arow + k + 16 * hi);
  av.h[1] = *(const v8bf*)(arow + k + 16 * hi + 8);
  return av.v;
}
// B fragment (32x16 bf16): lane L holds col N=L&15, K = 16*(L>>4)..+16, which
// for C = A*W^T is a contiguous run of row n of row-major W[N,K].
__device__ __forceinline__ v16bf load_b_frag(const unsigned short* wrow, int k, int hi) {
  return *(const v16bf*)(wrow + k + 16 * hi);
}

// One fused-K GEMM panel: dual accumulators break the WMMA D->C RAW chain.
__device__ __forceinline__ void gemm_part(const unsigned short* __restrict__ arow,
                                          const unsigned short* __restrict__ wrow,
                                          int K, int hi, v8f& acc0, v8f& acc1) {
  for (int k = 0; k < K; k += 64) {
    v16bf a0 = load_a_frag(arow, k, hi);
    v16bf b0 = load_b_frag(wrow, k, hi);
    v16bf a1 = load_a_frag(arow, k + 32, hi);
    v16bf b1 = load_b_frag(wrow, k + 32, hi);
    __builtin_prefetch((const void*)(wrow + k + 512), 0, 1);
    acc0 = __builtin_amdgcn_wmma_f32_16x16x32_bf16(false, a0, false, b0,
                                                   (short)0, acc0, false, false);
    acc1 = __builtin_amdgcn_wmma_f32_16x16x32_bf16(false, a1, false, b1,
                                                   (short)0, acc1, false, false);
  }
}

// ---------------------------------------------------------------------------
// Init: cast weights f32->bf16, precompute alpha = exp(-dt/tau_m), zero state.
// ---------------------------------------------------------------------------
__global__ void srnn_init(const float* __restrict__ Wi2h, const float* __restrict__ Wh2h,
                          const float* __restrict__ Wh2d, const float* __restrict__ Wd2d,
                          const float* __restrict__ tau_m_h, const float* __restrict__ tau_m_d,
                          unsigned short* __restrict__ wi, unsigned short* __restrict__ wh,
                          unsigned short* __restrict__ wd, unsigned short* __restrict__ wr,
                          float* __restrict__ alpha_h, float* __restrict__ alpha_d,
                          float* __restrict__ hm, float* __restrict__ dm,
                          unsigned short* __restrict__ hs0, unsigned short* __restrict__ ds0) {
  int tid = blockIdx.x * blockDim.x + threadIdx.x;
  int stride = gridDim.x * blockDim.x;
  for (int i = tid; i < H_ * I_; i += stride) wi[i] = f2bf_u16(Wi2h[i]);
  for (int i = tid; i < H_ * H_; i += stride) wh[i] = f2bf_u16(Wh2h[i]);
  for (int i = tid; i < D_ * H_; i += stride) wd[i] = f2bf_u16(Wh2d[i]);
  for (int i = tid; i < D_ * D_; i += stride) wr[i] = f2bf_u16(Wd2d[i]);
  for (int i = tid; i < H_; i += stride) alpha_h[i] = __expf(-1.0f / tau_m_h[i]);
  for (int i = tid; i < D_; i += stride) alpha_d[i] = __expf(-1.0f / tau_m_d[i]);
  for (int i = tid; i < B_ * H_; i += stride) { hm[i] = 0.0f; hs0[i] = 0; }
  for (int i = tid; i < B_ * D_; i += stride) { dm[i] = 0.0f; ds0[i] = 0; }
}

// ---------------------------------------------------------------------------
// Hidden layer step: h_in = [x_t | hs_prev] @ [W_i2h | W_h2h]^T + biases,
// then LIF update -> hm (in place), hs_next (bf16 spikes).
// Grid: (B/16) x (H/128), 256 threads = 8 waves, one 16x16 tile per wave.
// ---------------------------------------------------------------------------
__global__ __launch_bounds__(256)
void srnn_step_h(const float* __restrict__ x,
                 const unsigned short* __restrict__ Wi,   // [H_,I_] bf16
                 const unsigned short* __restrict__ Wh,   // [H_,H_] bf16
                 const float* __restrict__ bi, const float* __restrict__ bh,
                 const float* __restrict__ alpha,
                 float* __restrict__ hm,
                 const unsigned short* __restrict__ hs_prev,
                 unsigned short* __restrict__ hs_next,
                 int t) {
  __shared__ unsigned short sA[16 * 1536];               // 48 KB activation panel
  const int b0 = blockIdx.x * 16;
  const int tid = threadIdx.x;

  // Stage hs_prev rows (bf16 copy) into sA[r][512:1536]
#if ASYNC_LDS
  for (int e = tid; e < 16 * (H_ / 8); e += 256) {
    int r = e >> 7, k8 = e & 127;
    __builtin_amdgcn_global_load_async_to_lds_b128(
        (gv4p)&hs_prev[(size_t)(b0 + r) * H_ + k8 * 8],
        (sv4p)&sA[r * 1536 + I_ + k8 * 8], 0, 0);
  }
#else
  for (int e = tid; e < 16 * (H_ / 8); e += 256) {
    int r = e >> 7, k8 = e & 127;
    *(uint4*)&sA[r * 1536 + I_ + k8 * 8] =
        *(const uint4*)&hs_prev[(size_t)(b0 + r) * H_ + k8 * 8];
  }
#endif
  // Stage x_t rows (f32 -> bf16 convert) into sA[r][0:512]
  for (int e = tid; e < 16 * I_; e += 256) {
    int r = e >> 9, i = e & (I_ - 1);
    sA[r * 1536 + i] = f2bf_u16(x[((size_t)(b0 + r) * T_ + t) * I_ + i]);
  }
#if ASYNC_LDS
  __builtin_amdgcn_s_wait_asynccnt(0);
#endif
  __syncthreads();

  const int lane = tid & 31, wv = tid >> 5;
  const int lr = lane & 15, hi = lane >> 4;
  const int n0 = blockIdx.y * 128 + wv * 16;
  const unsigned short* sArow = &sA[lr * 1536];
  v8f acc0 = {}, acc1 = {};

  gemm_part(sArow,      Wi + (size_t)(n0 + lr) * I_, I_, hi, acc0, acc1);  // x part
  gemm_part(sArow + I_, Wh + (size_t)(n0 + lr) * H_, H_, hi, acc0, acc1);  // recurrent

  // LIF epilogue. C/D layout: VGPR v -> row v + 8*hi, col = lane&15.
  const int col = n0 + lr;
  const float al = alpha[col];
  const float bias = bi[col] + bh[col];
#pragma unroll
  for (int v = 0; v < 8; ++v) {
    int b = b0 + v + 8 * hi;
    size_t idx = (size_t)b * H_ + col;
    float h_in = acc0[v] + acc1[v] + bias;
    float mem = hm[idx] * al + (1.0f - al) * h_in - BJ0 * bf2f(hs_prev[idx]);
    hm[idx] = mem;
    hs_next[idx] = (mem - BJ0 > 0.0f) ? (unsigned short)0x3F80 : (unsigned short)0;
  }
}

// ---------------------------------------------------------------------------
// Dense layer step: d_in = [hs_new | ds_prev] @ [W_h2d | W_d2d]^T + biases,
// LIF update -> dm, ds_next, write spikes to out[:, t, :].
// Grid: (B/16) x (D/128), 256 threads.
// ---------------------------------------------------------------------------
__global__ __launch_bounds__(256)
void srnn_step_d(const unsigned short* __restrict__ hs,  // [B,H] fresh spikes
                 const unsigned short* __restrict__ Wd,  // [D_,H_] bf16
                 const unsigned short* __restrict__ Wr,  // [D_,D_] bf16
                 const float* __restrict__ b1, const float* __restrict__ b2,
                 const float* __restrict__ alpha,
                 float* __restrict__ dm,
                 const unsigned short* __restrict__ ds_prev,
                 unsigned short* __restrict__ ds_next,
                 float* __restrict__ out,                 // [B,T,D]
                 int t) {
  __shared__ unsigned short sA[16 * 1536];
  const int b0 = blockIdx.x * 16;
  const int tid = threadIdx.x;

#if ASYNC_LDS
  for (int e = tid; e < 16 * (H_ / 8); e += 256) {       // hs panel -> [0:1024)
    int r = e >> 7, k8 = e & 127;
    __builtin_amdgcn_global_load_async_to_lds_b128(
        (gv4p)&hs[(size_t)(b0 + r) * H_ + k8 * 8],
        (sv4p)&sA[r * 1536 + k8 * 8], 0, 0);
  }
  for (int e = tid; e < 16 * (D_ / 8); e += 256) {       // ds panel -> [1024:1536)
    int r = e >> 6, k8 = e & 63;
    __builtin_amdgcn_global_load_async_to_lds_b128(
        (gv4p)&ds_prev[(size_t)(b0 + r) * D_ + k8 * 8],
        (sv4p)&sA[r * 1536 + H_ + k8 * 8], 0, 0);
  }
  __builtin_amdgcn_s_wait_asynccnt(0);
#else
  for (int e = tid; e < 16 * (H_ / 8); e += 256) {       // hs panel -> [0:1024)
    int r = e >> 7, k8 = e & 127;
    *(uint4*)&sA[r * 1536 + k8 * 8] =
        *(const uint4*)&hs[(size_t)(b0 + r) * H_ + k8 * 8];
  }
  for (int e = tid; e < 16 * (D_ / 8); e += 256) {       // ds panel -> [1024:1536)
    int r = e >> 6, k8 = e & 63;
    *(uint4*)&sA[r * 1536 + H_ + k8 * 8] =
        *(const uint4*)&ds_prev[(size_t)(b0 + r) * D_ + k8 * 8];
  }
#endif
  __syncthreads();

  const int lane = tid & 31, wv = tid >> 5;
  const int lr = lane & 15, hi = lane >> 4;
  const int n0 = blockIdx.y * 128 + wv * 16;
  const unsigned short* sArow = &sA[lr * 1536];
  v8f acc0 = {}, acc1 = {};

  gemm_part(sArow,      Wd + (size_t)(n0 + lr) * H_, H_, hi, acc0, acc1);  // hidden
  gemm_part(sArow + H_, Wr + (size_t)(n0 + lr) * D_, D_, hi, acc0, acc1);  // recurrent

  const int col = n0 + lr;
  const float al = alpha[col];
  const float bias = b1[col] + b2[col];
#pragma unroll
  for (int v = 0; v < 8; ++v) {
    int b = b0 + v + 8 * hi;
    size_t idx = (size_t)b * D_ + col;
    float d_in = acc0[v] + acc1[v] + bias;
    float mem = dm[idx] * al + (1.0f - al) * d_in - BJ0 * bf2f(ds_prev[idx]);
    dm[idx] = mem;
    bool spk = (mem - BJ0 > 0.0f);
    ds_next[idx] = spk ? (unsigned short)0x3F80 : (unsigned short)0;
    out[((size_t)b * T_ + t) * D_ + col] = spk ? 1.0f : 0.0f;
  }
}

// Final hidden spikes -> f32 tail of output.
__global__ void srnn_final(const unsigned short* __restrict__ hs, float* __restrict__ out) {
  int i = blockIdx.x * blockDim.x + threadIdx.x;
  if (i < B_ * H_) out[i] = bf2f(hs[i]);
}

// ---------------------------------------------------------------------------
extern "C" void kernel_launch(void* const* d_in, const int* in_sizes, int n_in,
                              void* d_out, int out_size, void* d_ws, size_t ws_size,
                              hipStream_t stream) {
  (void)in_sizes; (void)n_in; (void)out_size; (void)ws_size;
  const float* x      = (const float*)d_in[0];
  const float* W_i2h  = (const float*)d_in[1];
  const float* b_i2h  = (const float*)d_in[2];
  const float* W_h2h  = (const float*)d_in[3];
  const float* b_h2h  = (const float*)d_in[4];
  const float* W_h2d  = (const float*)d_in[5];
  const float* b_h2d  = (const float*)d_in[6];
  const float* W_d2d  = (const float*)d_in[7];
  const float* b_d2d  = (const float*)d_in[8];
  // d_in[9]=tau_adp_h, d_in[11]=tau_adp_d: dead in forward (isAdapt=False).
  const float* tau_m_h = (const float*)d_in[10];
  const float* tau_m_d = (const float*)d_in[12];

  char* ws = (char*)d_ws;
  size_t off = 0;
  auto alloc = [&](size_t bytes) -> char* {
    char* p = ws + off;
    off = (off + bytes + 255) & ~(size_t)255;
    return p;
  };
  unsigned short* wi = (unsigned short*)alloc((size_t)H_ * I_ * 2);
  unsigned short* wh = (unsigned short*)alloc((size_t)H_ * H_ * 2);
  unsigned short* wd = (unsigned short*)alloc((size_t)D_ * H_ * 2);
  unsigned short* wr = (unsigned short*)alloc((size_t)D_ * D_ * 2);
  float* alpha_h = (float*)alloc(H_ * 4);
  float* alpha_d = (float*)alloc(D_ * 4);
  float* hm = (float*)alloc((size_t)B_ * H_ * 4);
  float* dm = (float*)alloc((size_t)B_ * D_ * 4);
  unsigned short* hsbuf[2] = {(unsigned short*)alloc((size_t)B_ * H_ * 2),
                              (unsigned short*)alloc((size_t)B_ * H_ * 2)};
  unsigned short* dsbuf[2] = {(unsigned short*)alloc((size_t)B_ * D_ * 2),
                              (unsigned short*)alloc((size_t)B_ * D_ * 2)};
  float* out = (float*)d_out;

  srnn_init<<<1024, 256, 0, stream>>>(W_i2h, W_h2h, W_h2d, W_d2d, tau_m_h, tau_m_d,
                                      wi, wh, wd, wr, alpha_h, alpha_d,
                                      hm, dm, hsbuf[0], dsbuf[0]);

  for (int t = 0; t < T_; ++t) {
    const unsigned short* hsp = hsbuf[t & 1];
    unsigned short* hsn = hsbuf[(t + 1) & 1];
    const unsigned short* dsp = dsbuf[t & 1];
    unsigned short* dsn = dsbuf[(t + 1) & 1];
    srnn_step_h<<<dim3(B_ / 16, H_ / 128), 256, 0, stream>>>(
        x, wi, wh, b_i2h, b_h2h, alpha_h, hm, hsp, hsn, t);
    srnn_step_d<<<dim3(B_ / 16, D_ / 128), 256, 0, stream>>>(
        hsn, wd, wr, b_h2d, b_d2d, alpha_d, dm, dsp, dsn, out, t);
  }

  srnn_final<<<(B_ * H_ + 255) / 256, 256, 0, stream>>>(
      hsbuf[T_ & 1], out + (size_t)B_ * T_ * D_);
}
